// DL_alignment_38594576121986
// MI455X (gfx1250) — matrix operand order repaired and checked by reading, not compiled, not run on hardware
//
#include <hip/hip_runtime.h>

// ---------------------------------------------------------------------------
// DL_alignment for MI455X (gfx1250): WMMA implicit-GEMM convs + fused
// cosine-similarity GEMM with packed atomic column max/argmax.
// B=2, C=64, H=W=192. f16 tensor-core path, f32 accumulate.
// conv: A = 2x global_load_b128, B = 2x ds_load_b128 ([n][k] LDS tile).
// gemm: LDS-free; both operands row-major [L][9C] -> all fragments are
//       16B global_load_b128 pairs at k+8h, k+16+8h. global_prefetch on next K.
// ---------------------------------------------------------------------------

typedef __attribute__((ext_vector_type(16))) _Float16 v16h;
typedef __attribute__((ext_vector_type(8)))  float    v8f;
typedef _Float16 half_t;

union V16 {            // 32-byte fragment viewed as halves or 2x uint4
  uint4 q[2];
  v16h  v;
};
union H16 {            // staging pack: 16 halves -> 2x uint4
  half_t h[16];
  uint4  q[2];
};

// ---------------------------------------------------------------- cast f32->f16
__global__ void cast_f16_kernel(const float* __restrict__ in,
                                half_t* __restrict__ out, int n) {
  int i = blockIdx.x * blockDim.x + threadIdx.x;
  if (i < n) out[i] = (half_t)in[i];
}

// ---------------------------------------------------------------- conv3 (WMMA)
// Implicit GEMM: O[m,n] = sum_k W[m,k] * patch[k,n], k = ci*9 + ky*3 + kx.
// X = concat(X1[C1], X2[C2]) channels (X2 may be null with C2=0).
// Block: 256 threads = 8 waves; tile = 32 (Cout) x 128 (pixels).
// Wave (wm, wn) in 2x4 owns a 16x32 tile => 2 WMMA per K-step, A reused.
// Wt rows must be allocated >= mbase+32 rows (padded; extra rows unused).
__global__ __launch_bounds__(256) void conv3_wmma_kernel(
    const half_t* __restrict__ X1, int C1,
    const half_t* __restrict__ X2, int C2,
    const half_t* __restrict__ Wt,      // [>=ceil32(Cout)][K] f16 row-major
    const float*  __restrict__ bias,    // [Cout] or null
    float*  __restrict__ outF,          // [B][Cout][H][W] or null
    half_t* __restrict__ outH,          // same, f16, or null
    int Cout, int H, int Wd) {
  const int HWp = H * Wd;
  const int b = blockIdx.z;
  const int nbase = blockIdx.x * 128;
  const int mbase = blockIdx.y * 32;
  const int K = (C1 + C2) * 9;
  const int tid  = threadIdx.x;
  const int lane = tid & 31;
  const int wave = tid >> 5;
  const int wm = wave >> 2;     // 0..1
  const int wn = wave & 3;      // 0..3, owns pixels [wn*32, wn*32+32)
  const int hl = lane >> 4;     // half-lane group

  __shared__ half_t Bs[128][32];   // [n][k], rows 64B

  v8f acc0 = {}, acc1 = {};

  // A fragment source: row m contiguous; halves i<8 -> k+0..7, i>=8 -> k+16..23,
  // upper half-lanes shifted by 8 (ISA 7.12.2 16-bit A layout).
  const int am = mbase + wm * 16 + (lane & 15);
  const half_t* aptr = Wt + (size_t)am * K + hl * 8;

  // staging coords (loop-invariant): this thread fills Bs[n_loc][k0..k0+15]
  const int n_loc = tid & 127;
  const int k0    = (tid >> 7) * 16;
  const int npix  = nbase + n_loc;
  const int h0 = npix / Wd, w0 = npix % Wd;

  for (int kt = 0; kt < K; kt += 32) {
    // ---- stage B tile (branchless gather, packed 2x b128 store)
    H16 tmp;
#pragma unroll
    for (int j = 0; j < 16; ++j) {
      int kg = kt + k0 + j;
      int ci = kg / 9;
      int rr = kg - ci * 9;
      int ky = rr / 3 - 1;
      int kx = rr - (rr / 3) * 3 - 1;
      int hh = h0 + ky, ww = w0 + kx;
      bool inb = ((unsigned)hh < (unsigned)H) && ((unsigned)ww < (unsigned)Wd);
      int hc = inb ? hh : 0, wc = inb ? ww : 0;
      const half_t* sp = (ci < C1)
          ? X1 + ((size_t)b * C1 + ci) * HWp
          : X2 + ((size_t)b * C2 + (ci - C1)) * HWp;
      half_t v = sp[hc * Wd + wc];
      tmp.h[j] = inb ? v : (half_t)0.f;
    }
    *(uint4*)&Bs[n_loc][k0]     = tmp.q[0];
    *(uint4*)&Bs[n_loc][k0 + 8] = tmp.q[1];
    __syncthreads();

    // ---- fragments: A 2x global b128, B 2x ds b128 per subtile
    V16 af;
    af.q[0] = *(const uint4*)(aptr + kt);
    af.q[1] = *(const uint4*)(aptr + kt + 16);
    __builtin_prefetch(aptr + kt + 32, 0, 1);   // next K-block (speculative)
    const int kb2 = hl * 8;    // B layout: halves i<8 -> k = i + hl*8, i>=8 -> k = 16 + hl*8 + (i-8)
    V16 bf0, bf1;
    {
      const half_t* p0 = &Bs[wn * 32 + (lane & 15)][kb2];
      bf0.q[0] = *(const uint4*)p0;
      bf0.q[1] = *(const uint4*)(p0 + 16);
      const half_t* p1 = &Bs[wn * 32 + 16 + (lane & 15)][kb2];
      bf1.q[0] = *(const uint4*)p1;
      bf1.q[1] = *(const uint4*)(p1 + 16);
    }
    acc0 = __builtin_amdgcn_wmma_f32_16x16x32_f16(
        false, af.v, false, bf0.v, (short)0, acc0, false, false);
    acc1 = __builtin_amdgcn_wmma_f32_16x16x32_f16(
        false, af.v, false, bf1.v, (short)0, acc1, false, false);
    __syncthreads();
  }

  // ---- epilogue: bias + leaky ReLU, dual-precision store
  int n0 = nbase + wn * 32 + (lane & 15);
  int n1 = n0 + 16;
#pragma unroll
  for (int r = 0; r < 8; ++r) {
    int m = mbase + wm * 16 + r + hl * 8;
    if (m < Cout) {
      float bsv = bias ? bias[m] : 0.f;
      float v0 = acc0[r] + bsv;
      float v1 = acc1[r] + bsv;
      v0 = (v0 >= 0.f) ? v0 : 0.1f * v0;
      v1 = (v1 >= 0.f) ? v1 : 0.1f * v1;
      size_t base = ((size_t)(b * Cout + m)) * HWp;
      if (outF) { outF[base + n0] = v0; outF[base + n1] = v1; }
      if (outH) { outH[base + n0] = (half_t)v0; outH[base + n1] = (half_t)v1; }
    }
  }
}

// ------------------------------------------- cosine GEMM + fused col max/argmax
// R[l,m] = dot(Trow_l, Rrow_m); both operands stored row-major [B][L][K].
// LDS-free: every fragment is 2x contiguous global_load_b128.
// Block: 8 waves; wave = one 16-row tile x two 16-col tiles (A reused).
// Column max over l tracked via packed (orderable-key<<32 | row) u64 atomicMax.
__global__ __launch_bounds__(256) void gemm_colmax_kernel(
    const half_t* __restrict__ Amat,   // [B][L][K] t-patch rows (R rows l)
    const half_t* __restrict__ Bmat,   // [B][L][K] r-patch rows (R cols m)
    unsigned long long* __restrict__ packed,  // [B][L], pre-zeroed
    int L, int K) {
  const int b = blockIdx.z;
  const int nbase = blockIdx.x * 32;
  const int mbase = blockIdx.y * 128;
  const int lane = threadIdx.x & 31;
  const int wave = threadIdx.x >> 5;
  const int hl   = lane >> 4;

  v8f acc0 = {}, acc1 = {};
  const int am = mbase + wave * 16 + (lane & 15);
  const half_t* aptr = Amat + ((size_t)b * L + am) * K + hl * 8;
  const int bn = nbase + (lane & 15);
  const half_t* bptr0 = Bmat + ((size_t)b * L + bn) * K + hl * 8;
  const half_t* bptr1 = Bmat + ((size_t)b * L + bn + 16) * K + hl * 8;

  for (int kt = 0; kt < K; kt += 32) {
    V16 af, bf0, bf1;
    af.q[0]  = *(const uint4*)(aptr + kt);
    af.q[1]  = *(const uint4*)(aptr + kt + 16);
    bf0.q[0] = *(const uint4*)(bptr0 + kt);
    bf0.q[1] = *(const uint4*)(bptr0 + kt + 16);
    bf1.q[0] = *(const uint4*)(bptr1 + kt);
    bf1.q[1] = *(const uint4*)(bptr1 + kt + 16);
    __builtin_prefetch(aptr + kt + 32, 0, 1);
    __builtin_prefetch(bptr0 + kt + 32, 0, 1);
    __builtin_prefetch(bptr1 + kt + 32, 0, 1);
    acc0 = __builtin_amdgcn_wmma_f32_16x16x32_f16(
        false, af.v, false, bf0.v, (short)0, acc0, false, false);
    acc1 = __builtin_amdgcn_wmma_f32_16x16x32_f16(
        false, af.v, false, bf1.v, (short)0, acc1, false, false);
  }

  float best0 = -1e30f, best1 = -1e30f;
  int row0 = 0, row1 = 0;
#pragma unroll
  for (int r = 0; r < 8; ++r) {
    int row = mbase + wave * 16 + r + hl * 8;
    if (acc0[r] > best0) { best0 = acc0[r]; row0 = row; }
    if (acc1[r] > best1) { best1 = acc1[r]; row1 = row; }
  }
  unsigned int fb0 = __float_as_uint(best0);
  unsigned int k0  = (fb0 & 0x80000000u) ? ~fb0 : (fb0 | 0x80000000u);
  unsigned int fb1 = __float_as_uint(best1);
  unsigned int k1  = (fb1 & 0x80000000u) ? ~fb1 : (fb1 | 0x80000000u);
  atomicMax(packed + (size_t)b * L + bn,
            ((unsigned long long)k0 << 32) | (unsigned)row0);
  atomicMax(packed + (size_t)b * L + bn + 16,
            ((unsigned long long)k1 << 32) | (unsigned)row1);
}

__global__ void unpack_max_kernel(const unsigned long long* __restrict__ packed,
                                  float* __restrict__ Rstar,
                                  int* __restrict__ arg, int n) {
  int i = blockIdx.x * blockDim.x + threadIdx.x;
  if (i >= n) return;
  unsigned long long pk = packed[i];
  unsigned int key = (unsigned int)(pk >> 32);
  unsigned int fb  = (key & 0x80000000u) ? (key & 0x7FFFFFFFu) : ~key;
  Rstar[i] = __uint_as_float(fb);
  arg[i]   = (int)(pk & 0xFFFFFFFFull);
}

// ------------------------------------------------------- deformable conv (VALU)
__global__ void deform_conv_kernel(const float* __restrict__ X,     // [B][64][H][W]
                                   const float* __restrict__ Off,   // [B][18][H][W]
                                   const float* __restrict__ Wdf,   // [64][16][9]
                                   half_t* __restrict__ outH,       // lrelu, f16
                                   int H, int Wd, int Btot) {
  const int HWp = H * Wd;
  int idx = blockIdx.x * blockDim.x + threadIdx.x;
  if (idx >= Btot * 4 * HWp) return;
  int pix = idx % HWp;
  int bg  = idx / HWp;
  int g   = bg & 3;
  int b   = bg >> 2;
  int h = pix / Wd, w = pix % Wd;

  float acc[16];
#pragma unroll
  for (int o = 0; o < 16; ++o) acc[o] = 0.f;

  for (int p = 0; p < 9; ++p) {
    float dy = Off[((size_t)(b * 18 + 2 * p)) * HWp + pix];
    float dx = Off[((size_t)(b * 18 + 2 * p + 1)) * HWp + pix];
    float py = (float)(h + p / 3 - 1) + dy;
    float px = (float)(w + p % 3 - 1) + dx;
    float y0f = floorf(py), x0f = floorf(px);
    int y0 = (int)y0f, x0 = (int)x0f;
    float wy = py - y0f, wx = px - x0f;
    float w00 = (1.f - wy) * (1.f - wx), w01 = (1.f - wy) * wx;
    float w10 = wy * (1.f - wx),         w11 = wy * wx;
    bool v00 = (y0 >= 0 && y0 < H     && x0 >= 0     && x0 < Wd);
    bool v01 = (y0 >= 0 && y0 < H     && x0 + 1 >= 0 && x0 + 1 < Wd);
    bool v10 = (y0 + 1 >= 0 && y0 + 1 < H && x0 >= 0     && x0 < Wd);
    bool v11 = (y0 + 1 >= 0 && y0 + 1 < H && x0 + 1 >= 0 && x0 + 1 < Wd);
    for (int ci = 0; ci < 16; ++ci) {
      const float* xc = X + ((size_t)(b * 64 + g * 16 + ci)) * HWp;
      float s = 0.f;
      if (v00) s += w00 * xc[y0 * Wd + x0];
      if (v01) s += w01 * xc[y0 * Wd + x0 + 1];
      if (v10) s += w10 * xc[(y0 + 1) * Wd + x0];
      if (v11) s += w11 * xc[(y0 + 1) * Wd + x0 + 1];
#pragma unroll
      for (int o = 0; o < 16; ++o)
        acc[o] += s * Wdf[((size_t)((g * 16 + o) * 16 + ci)) * 9 + p];
    }
  }
#pragma unroll
  for (int o = 0; o < 16; ++o) {
    float v = acc[o];
    v = (v >= 0.f) ? v : 0.1f * v;
    outH[((size_t)(b * 64 + g * 16 + o)) * HWp + pix] = (half_t)v;
  }
}

// ------------------------------------------------------- attention (dot+sigmoid)
__global__ void attention_kernel(const float* __restrict__ Q,
                                 const float* __restrict__ Kt,
                                 float* __restrict__ att, int HWp, int Btot) {
  int idx = blockIdx.x * blockDim.x + threadIdx.x;
  if (idx >= Btot * HWp) return;
  int b = idx / HWp, pix = idx % HWp;
  float s = 0.f;
  for (int c = 0; c < 64; ++c)
    s += Q[((size_t)(b * 64 + c)) * HWp + pix] *
         Kt[((size_t)(b * 64 + c)) * HWp + pix];
  att[idx] = 1.f / (1.f + expf(-s));
}

// ------------------------------------------- per-pixel broadcast multiply -> f16
__global__ void mul_bcast_f16_kernel(const float* __restrict__ X,   // [B][C][HW]
                                     const float* __restrict__ S,   // [B][HW]
                                     half_t* __restrict__ out, int Cc, int HWp,
                                     int total) {
  int idx = blockIdx.x * blockDim.x + threadIdx.x;
  if (idx >= total) return;
  int b   = idx / (Cc * HWp);
  int pix = idx % HWp;
  out[idx] = (half_t)(X[idx] * S[(size_t)b * HWp + pix]);
}

// --------------------------------------------- bilinear resize, align_corners=1
__global__ void resize_ac_kernel(const float* __restrict__ in,
                                 float* __restrict__ out, int Cc,
                                 int Hin, int Win, int Hout, int Wout,
                                 int Btot) {
  int idx = blockIdx.x * blockDim.x + threadIdx.x;
  int total = Btot * Cc * Hout * Wout;
  if (idx >= total) return;
  int x  = idx % Wout;
  int y  = (idx / Wout) % Hout;
  int bc = idx / (Wout * Hout);
  float sy = (Hout > 1) ? (float)(Hin - 1) / (float)(Hout - 1) : 0.f;
  float sx = (Wout > 1) ? (float)(Win - 1) / (float)(Wout - 1) : 0.f;
  float cy = y * sy, cx = x * sx;
  int y0 = (int)floorf(cy); y0 = y0 > Hin - 2 ? Hin - 2 : (y0 < 0 ? 0 : y0);
  int x0 = (int)floorf(cx); x0 = x0 > Win - 2 ? Win - 2 : (x0 < 0 ? 0 : x0);
  float wy = cy - (float)y0, wx = cx - (float)x0;
  const float* p = in + (size_t)bc * Hin * Win;
  float a = p[y0 * Win + x0],       b1 = p[y0 * Win + x0 + 1];
  float c = p[(y0 + 1) * Win + x0], d  = p[(y0 + 1) * Win + x0 + 1];
  float top = a + (b1 - a) * wx, bot = c + (d - c) * wx;
  out[idx] = top + (bot - top) * wy;
}

// ------------------------------- unfold(k=3,p=1,s=1) + L2-normalize over 9C dim
// thread per (b, l); writes patch-major rows [B][L][9C] f16.
__global__ void unfold_norm_kernel(const float* __restrict__ Xd,  // [B][C][Hs][Ws]
                                   half_t* __restrict__ outT,     // [B][L][9C]
                                   int Cc, int Hs, int Ws, int Btot) {
  int L = Hs * Ws;
  int idx = blockIdx.x * blockDim.x + threadIdx.x;
  if (idx >= Btot * L) return;
  int b = idx / L, l = idx % L;
  int ly = l / Ws, lx = l % Ws;
  float ss = 0.f;
  for (int c = 0; c < Cc; ++c)
    for (int p = 0; p < 9; ++p) {
      int yy = ly + p / 3 - 1, xx = lx + p % 3 - 1;
      float v = (yy >= 0 && yy < Hs && xx >= 0 && xx < Ws)
                    ? Xd[((size_t)(b * Cc + c) * Hs + yy) * Ws + xx] : 0.f;
      ss += v * v;
    }
  float inv = 1.f / fmaxf(sqrtf(ss), 1e-12f);
  int K9 = Cc * 9;
  for (int c = 0; c < Cc; ++c)
    for (int p = 0; p < 9; ++p) {
      int yy = ly + p / 3 - 1, xx = lx + p % 3 - 1;
      float v = (yy >= 0 && yy < Hs && xx >= 0 && xx < Ws)
                    ? Xd[((size_t)(b * Cc + c) * Hs + yy) * Ws + xx] : 0.f;
      outT[((size_t)b * L + l) * K9 + c * 9 + p] = (half_t)(v * inv);
    }
}

// -------- fold(unfold(T,12,4,4) gathered by argmax)/9 as a pure gather kernel
__global__ void hard_fold_kernel(const float* __restrict__ Tref,  // [B][64][H][W]
                                 const int* __restrict__ arg,     // [B][2304]
                                 half_t* __restrict__ outH,       // [B][64][H][W]
                                 int H, int Wd, int Btot) {
  int HWp = H * Wd;
  int idx = blockIdx.x * blockDim.x + threadIdx.x;
  if (idx >= Btot * 64 * HWp) return;
  int x  = idx % Wd;
  int y  = (idx / Wd) % H;
  int c  = (idx / HWp) % 64;
  int b  = idx / (64 * HWp);
  int iy_lo = (y - 4) / 4; if (iy_lo < 0) iy_lo = 0;
  int iy_hi = (y + 4) / 4; if (iy_hi > 47) iy_hi = 47;
  int ix_lo = (x - 4) / 4; if (ix_lo < 0) ix_lo = 0;
  int ix_hi = (x + 4) / 4; if (ix_hi > 47) ix_hi = 47;
  float sum = 0.f;
  for (int iy = iy_lo; iy <= iy_hi; ++iy)
    for (int ix = ix_lo; ix <= ix_hi; ++ix) {
      int m  = iy * 48 + ix;
      int am = arg[(size_t)b * 2304 + m];
      int py0 = 4 * (am / 48) - 4, px0 = 4 * (am % 48) - 4;
      int sy = py0 + (y - (4 * iy - 4));
      int sx = px0 + (x - (4 * ix - 4));
      if (sy >= 0 && sy < H && sx >= 0 && sx < Wd)
        sum += Tref[((size_t)(b * 64 + c) * H + sy) * Wd + sx];
    }
  outH[idx] = (half_t)(sum / 9.f);
}

// ===========================================================================
extern "C" void kernel_launch(void* const* d_in, const int* in_sizes, int n_in,
                              void* d_out, int out_size, void* d_ws, size_t ws_size,
                              hipStream_t stream) {
  (void)in_sizes; (void)n_in; (void)out_size; (void)ws_size;
  const int Bn = 2, Cc = 64, Hh = 192, Ww = 192;
  const int HWp = Hh * Ww;                       // 36864
  const size_t NCHW = (size_t)Bn * Cc * HWp;     // 4718592
  const int Hs = 48, L = Hs * Hs;                // 2304
  const int K9 = Cc * 9;                         // 576

  const float* in_rend  = (const float*)d_in[0];
  const float* in_wref0 = (const float*)d_in[1];
  const float* in_tref0 = (const float*)d_in[2];
  const float* in_wref1 = (const float*)d_in[3];
  const float* in_tref1 = (const float*)d_in[4];
  const float* w_of0 = (const float*)d_in[5];  const float* b_of0 = (const float*)d_in[6];
  const float* w_of1 = (const float*)d_in[7];  const float* b_of1 = (const float*)d_in[8];
  const float* w_df0 = (const float*)d_in[9];  const float* w_df1 = (const float*)d_in[10];
  const float* w_q  = (const float*)d_in[11];  const float* b_q  = (const float*)d_in[12];
  const float* w_k0 = (const float*)d_in[13];  const float* b_k0 = (const float*)d_in[14];
  const float* w_k1 = (const float*)d_in[15];  const float* b_k1 = (const float*)d_in[16];
  const float* w_v0 = (const float*)d_in[17];  const float* b_v0 = (const float*)d_in[18];
  const float* w_v1 = (const float*)d_in[19];  const float* b_v1 = (const float*)d_in[20];
  const float* w_f0 = (const float*)d_in[21];  const float* b_f0 = (const float*)d_in[22];
  const float* w_f1 = (const float*)d_in[23];  const float* b_f1 = (const float*)d_in[24];
  const float* w_fo0 = (const float*)d_in[25]; const float* b_fo0 = (const float*)d_in[26];
  const float* w_fo1 = (const float*)d_in[27]; const float* b_fo1 = (const float*)d_in[28];
  const float* w_ch0 = (const float*)d_in[29]; const float* b_ch0 = (const float*)d_in[30];
  const float* w_ch1 = (const float*)d_in[31]; const float* b_ch1 = (const float*)d_in[32];
  const float* w_o0 = (const float*)d_in[33];  const float* b_o0 = (const float*)d_in[34];
  const float* w_o1 = (const float*)d_in[35];  const float* b_o1 = (const float*)d_in[36];

  float* outp = (float*)d_out;
  float* out_fw0 = outp + 0 * NCHW;
  float* out_s0  = outp + 1 * NCHW;
  float* out_fw1 = outp + 2 * NCHW;
  float* out_s1  = outp + 3 * NCHW;

  // ---- workspace bump allocator (256B aligned)
  char* wsb = (char*)d_ws;
  size_t cur = 0;
  auto alloc = [&](size_t bytes) -> void* {
    void* p = wsb + cur;
    cur = (cur + bytes + 255) & ~(size_t)255;
    return p;
  };

  // f16 activation copies / conv inputs
  half_t* rend16  = (half_t*)alloc(NCHW * 2);
  half_t* wref016 = (half_t*)alloc(NCHW * 2);
  half_t* wref116 = (half_t*)alloc(NCHW * 2);
  half_t* Wr0_16  = (half_t*)alloc(NCHW * 2);
  half_t* Wr1_16  = (half_t*)alloc(NCHW * 2);
  half_t* VA0_16  = (half_t*)alloc(NCHW * 2);
  half_t* VA1_16  = (half_t*)alloc(NCHW * 2);
  half_t* f0_16   = (half_t*)alloc(NCHW * 2);
  half_t* f1_16   = (half_t*)alloc(NCHW * 2);
  half_t* Hard016 = (half_t*)alloc(NCHW * 2);
  half_t* Hard116 = (half_t*)alloc(NCHW * 2);
  half_t* hs0_16  = (half_t*)alloc(NCHW * 2);
  half_t* hs1_16  = (half_t*)alloc(NCHW * 2);
  // f32 intermediates
  float* off0 = (float*)alloc((size_t)Bn * 18 * HWp * 4);
  float* off1 = (float*)alloc((size_t)Bn * 18 * HWp * 4);
  float* Qb   = (float*)alloc(NCHW * 4);
  float* K0b  = (float*)alloc(NCHW * 4);
  float* K1b  = (float*)alloc(NCHW * 4);
  float* V0b  = (float*)alloc(NCHW * 4);
  float* V1b  = (float*)alloc(NCHW * 4);
  float* hf0  = (float*)alloc(NCHW * 4);
  float* hf1  = (float*)alloc(NCHW * 4);
  float* att0 = (float*)alloc((size_t)Bn * HWp * 4);
  float* att1 = (float*)alloc((size_t)Bn * HWp * 4);
  float* sm0  = (float*)alloc((size_t)Bn * HWp * 4);
  float* sm1  = (float*)alloc((size_t)Bn * HWp * 4);
  // tele (all patch matrices row-major [B][L][9C])
  float* T0d = (float*)alloc((size_t)Bn * Cc * L * 4);
  float* T1d = (float*)alloc((size_t)Bn * Cc * L * 4);
  float* rdb = (float*)alloc((size_t)Bn * Cc * L * 4);
  half_t* t0n = (half_t*)alloc((size_t)Bn * L * K9 * 2);
  half_t* t1n = (half_t*)alloc((size_t)Bn * L * K9 * 2);
  half_t* rnT = (half_t*)alloc((size_t)Bn * L * K9 * 2);
  unsigned long long* pk0 = (unsigned long long*)alloc((size_t)Bn * L * 8);
  unsigned long long* pk1 = (unsigned long long*)alloc((size_t)Bn * L * 8);
  float* Rs0 = (float*)alloc((size_t)Bn * L * 4);
  float* Rs1 = (float*)alloc((size_t)Bn * L * 4);
  int* arg0 = (int*)alloc((size_t)Bn * L * 4);
  int* arg1 = (int*)alloc((size_t)Bn * L * 4);
  // f16 weights (of-convs padded to 32 rows: A fragments load rows < mbase+32)
  half_t* wh_of0 = (half_t*)alloc((size_t)32 * 1152 * 2);
  half_t* wh_of1 = (half_t*)alloc((size_t)32 * 1152 * 2);
  half_t* wh_q  = (half_t*)alloc((size_t)64 * 576 * 2);
  half_t* wh_k0 = (half_t*)alloc((size_t)64 * 576 * 2);
  half_t* wh_k1 = (half_t*)alloc((size_t)64 * 576 * 2);
  half_t* wh_v0 = (half_t*)alloc((size_t)64 * 576 * 2);
  half_t* wh_v1 = (half_t*)alloc((size_t)64 * 576 * 2);
  half_t* wh_f0 = (half_t*)alloc((size_t)64 * 576 * 2);
  half_t* wh_f1 = (half_t*)alloc((size_t)64 * 576 * 2);
  half_t* wh_o0 = (half_t*)alloc((size_t)64 * 576 * 2);
  half_t* wh_o1 = (half_t*)alloc((size_t)64 * 576 * 2);
  half_t* wh_fo0 = (half_t*)alloc((size_t)64 * 1152 * 2);
  half_t* wh_fo1 = (half_t*)alloc((size_t)64 * 1152 * 2);
  half_t* wh_ch0 = (half_t*)alloc((size_t)64 * 1152 * 2);
  half_t* wh_ch1 = (half_t*)alloc((size_t)64 * 1152 * 2);

  auto cast = [&](const float* s, half_t* d, size_t n) {
    cast_f16_kernel<<<dim3((unsigned)((n + 255) / 256)), 256, 0, stream>>>(s, d, (int)n);
  };
  auto conv = [&](const half_t* X1, int C1, const half_t* X2, int C2,
                  const half_t* Wt, const float* bias, float* oF, half_t* oH,
                  int Cout) {
    dim3 g(HWp / 128, (Cout + 31) / 32, Bn);
    conv3_wmma_kernel<<<g, 256, 0, stream>>>(X1, C1, X2, C2, Wt, bias, oF, oH,
                                             Cout, Hh, Ww);
  };
  auto resize = [&](const float* s, float* d, int C_, int Hi, int Wi, int Ho, int Wo) {
    int tot = Bn * C_ * Ho * Wo;
    resize_ac_kernel<<<dim3((tot + 255) / 256), 256, 0, stream>>>(
        s, d, C_, Hi, Wi, Ho, Wo, Bn);
  };

  // ---- stage 0: f16 casts (activations + weights)
  cast(in_rend, rend16, NCHW);
  cast(in_wref0, wref016, NCHW);
  cast(in_wref1, wref116, NCHW);
  cast(w_of0, wh_of0, (size_t)18 * 1152);
  cast(w_of1, wh_of1, (size_t)18 * 1152);
  cast(w_q, wh_q, (size_t)64 * 576);   cast(w_k0, wh_k0, (size_t)64 * 576);
  cast(w_k1, wh_k1, (size_t)64 * 576); cast(w_v0, wh_v0, (size_t)64 * 576);
  cast(w_v1, wh_v1, (size_t)64 * 576); cast(w_f0, wh_f0, (size_t)64 * 576);
  cast(w_f1, wh_f1, (size_t)64 * 576); cast(w_o0, wh_o0, (size_t)64 * 576);
  cast(w_o1, wh_o1, (size_t)64 * 576);
  cast(w_fo0, wh_fo0, (size_t)64 * 1152); cast(w_fo1, wh_fo1, (size_t)64 * 1152);
  cast(w_ch0, wh_ch0, (size_t)64 * 1152); cast(w_ch1, wh_ch1, (size_t)64 * 1152);

  // ---- Wide_align
  conv(rend16, 64, wref016, 64, wh_of0, b_of0, off0, nullptr, 18);
  conv(rend16, 64, wref116, 64, wh_of1, b_of1, off1, nullptr, 18);
  {
    int tot = Bn * 4 * HWp;
    deform_conv_kernel<<<dim3((tot + 255) / 256), 256, 0, stream>>>(
        in_wref0, off0, w_df0, Wr0_16, Hh, Ww, Bn);
    deform_conv_kernel<<<dim3((tot + 255) / 256), 256, 0, stream>>>(
        in_wref1, off1, w_df1, Wr1_16, Hh, Ww, Bn);
  }
  conv(rend16, 64, nullptr, 0, wh_q, b_q, Qb, nullptr, 64);
  conv(Wr0_16, 64, nullptr, 0, wh_k0, b_k0, K0b, nullptr, 64);
  conv(Wr1_16, 64, nullptr, 0, wh_k1, b_k1, K1b, nullptr, 64);
  conv(Wr0_16, 64, nullptr, 0, wh_v0, b_v0, V0b, nullptr, 64);
  conv(Wr1_16, 64, nullptr, 0, wh_v1, b_v1, V1b, nullptr, 64);
  {
    int tot = Bn * HWp;
    attention_kernel<<<dim3((tot + 255) / 256), 256, 0, stream>>>(Qb, K0b, att0, HWp, Bn);
    attention_kernel<<<dim3((tot + 255) / 256), 256, 0, stream>>>(Qb, K1b, att1, HWp, Bn);
    int tot2 = (int)NCHW;
    mul_bcast_f16_kernel<<<dim3((tot2 + 255) / 256), 256, 0, stream>>>(
        V0b, att0, VA0_16, Cc, HWp, tot2);
    mul_bcast_f16_kernel<<<dim3((tot2 + 255) / 256), 256, 0, stream>>>(
        V1b, att1, VA1_16, Cc, HWp, tot2);
  }
  conv(VA0_16, 64, nullptr, 0, wh_f0, b_f0, nullptr, f0_16, 64);
  conv(VA1_16, 64, nullptr, 0, wh_f1, b_f1, nullptr, f1_16, 64);
  conv(f0_16, 64, rend16, 64, wh_fo0, b_fo0, out_fw0, nullptr, 64);
  conv(f1_16, 64, rend16, 64, wh_fo1, b_fo1, out_fw1, nullptr, 64);

  // ---- Tele_align
  resize(in_tref0, T0d, Cc, Hh, Ww, Hs, Hs);
  resize(in_tref1, T1d, Cc, Hh, Ww, Hs, Hs);
  resize(in_rend, rdb, Cc, Hh, Ww, Hs, Hs);
  {
    int tot = Bn * L;
    unfold_norm_kernel<<<dim3((tot + 255) / 256), 256, 0, stream>>>(
        rdb, rnT, Cc, Hs, Hs, Bn);
    unfold_norm_kernel<<<dim3((tot + 255) / 256), 256, 0, stream>>>(
        T0d, t0n, Cc, Hs, Hs, Bn);
    unfold_norm_kernel<<<dim3((tot + 255) / 256), 256, 0, stream>>>(
        T1d, t1n, Cc, Hs, Hs, Bn);
  }
  hipMemsetAsync(pk0, 0, (size_t)Bn * L * 8, stream);
  hipMemsetAsync(pk1, 0, (size_t)Bn * L * 8, stream);
  {
    dim3 g(L / 32, L / 128, Bn);   // 72 x 18 x 2
    gemm_colmax_kernel<<<g, 256, 0, stream>>>(t0n, rnT, pk0, L, K9);
    gemm_colmax_kernel<<<g, 256, 0, stream>>>(t1n, rnT, pk1, L, K9);
    int tot = Bn * L;
    unpack_max_kernel<<<dim3((tot + 255) / 256), 256, 0, stream>>>(pk0, Rs0, arg0, tot);
    unpack_max_kernel<<<dim3((tot + 255) / 256), 256, 0, stream>>>(pk1, Rs1, arg1, tot);
  }
  {
    int tot = (int)NCHW;
    hard_fold_kernel<<<dim3((tot + 255) / 256), 256, 0, stream>>>(
        in_tref0, arg0, Hard016, Hh, Ww, Bn);
    hard_fold_kernel<<<dim3((tot + 255) / 256), 256, 0, stream>>>(
        in_tref1, arg1, Hard116, Hh, Ww, Bn);
  }
  conv(rend16, 64, Hard016, 64, wh_ch0, b_ch0, hf0, nullptr, 64);
  conv(rend16, 64, Hard116, 64, wh_ch1, b_ch1, hf1, nullptr, 64);
  resize(Rs0, sm0, 1, Hs, Hs, Hh, Ww);
  resize(Rs1, sm1, 1, Hs, Hs, Hh, Ww);
  {
    int tot = (int)NCHW;
    mul_bcast_f16_kernel<<<dim3((tot + 255) / 256), 256, 0, stream>>>(
        hf0, sm0, hs0_16, Cc, HWp, tot);
    mul_bcast_f16_kernel<<<dim3((tot + 255) / 256), 256, 0, stream>>>(
        hf1, sm1, hs1_16, Cc, HWp, tot);
  }
  conv(hs0_16, 64, nullptr, 0, wh_o0, b_o0, out_s0, nullptr, 64);
  conv(hs1_16, 64, nullptr, 0, wh_o1, b_o1, out_s1, nullptr, 64);
}
